// UnsupervisedRouter_12120397709535
// MI455X (gfx1250) — matrix operand order repaired and checked by reading, not compiled, not run on hardware
//
#include <hip/hip_runtime.h>
#include <math.h>

typedef __attribute__((ext_vector_type(16))) __bf16 v16bf;
typedef __attribute__((ext_vector_type(8)))  float  v8f;

constexpr int TOKENS         = 16384;
constexpr int HID            = 4096;
constexpr int NEXP           = 64;
constexpr int ROWS_PER_BLOCK = 64;       // 4 waves * 16 rows
constexpr int BLOCK_THREADS  = 128;      // 4 wave32
constexpr int LDS_STRIDE     = NEXP + 1; // 65: bank-conflict-free rows AND columns
constexpr int W_ELEMS        = NEXP * HID;           // 262144
constexpr size_t WS_RED_BYTES = 1024;                // ws[0..64] floats + pad

// Convert 8 consecutive fp32 into bf16 hi/lo split parts at element offset BASE.
template <int BASE>
__device__ __forceinline__ void cvt8(const float* __restrict__ p, v16bf& hi, v16bf& lo) {
  float4 f0 = *(const float4*)(p);
  float4 f1 = *(const float4*)(p + 4);
  float fv[8] = {f0.x, f0.y, f0.z, f0.w, f1.x, f1.y, f1.z, f1.w};
#pragma unroll
  for (int i = 0; i < 8; ++i) {
    __bf16 h = (__bf16)fv[i];
    hi[BASE + i] = h;
    lo[BASE + i] = (__bf16)(fv[i] - (float)h);
  }
}

#define WMMA_BF16(C, A, B) \
  __builtin_amdgcn_wmma_f32_16x16x32_bf16(false, (A), false, (B), (short)0, (C), false, false)

__global__ void zero_ws_kernel(float* __restrict__ ws) {
  int t = threadIdx.x;
  if (t <= NEXP) ws[t] = 0.0f;
}

// One-time split of W (fp32) into bf16 hi/lo planes, row-major (so B fragments
// are contiguous 32-byte loads in the main kernel).
__global__ __launch_bounds__(256)
void prep_w_kernel(const float* __restrict__ W,
                   __bf16* __restrict__ whi, __bf16* __restrict__ wlo) {
  int i = (blockIdx.x * 256 + threadIdx.x) * 2;   // W_ELEMS/2 threads total
  float f0 = W[i], f1 = W[i + 1];
  __bf16 h0 = (__bf16)f0, h1 = (__bf16)f1;
  whi[i] = h0;  whi[i + 1] = h1;
  wlo[i]     = (__bf16)(f0 - (float)h0);
  wlo[i + 1] = (__bf16)(f1 - (float)h1);
}

__global__ __launch_bounds__(BLOCK_THREADS)
void router_kernel(const float* __restrict__ x,
                   const __bf16* __restrict__ whi, const __bf16* __restrict__ wlo,
                   float* __restrict__ out, float* __restrict__ ws) {
  __shared__ float lds[ROWS_PER_BLOCK * LDS_STRIDE];
  __shared__ float lds_spk[ROWS_PER_BLOCK];

  const int tid  = threadIdx.x;
  const int lane = tid & 31;
  const int wave = tid >> 5;
  const int half = lane >> 4;   // which 16-lane half of the wave
  const int ln   = lane & 15;

  const int rowBase = blockIdx.x * ROWS_PER_BLOCK + wave * 16;
  const float* __restrict__ xrow = x + (size_t)(rowBase + ln) * HID;
  // B fragment rows: lane = expert column of each 16-wide n-tile.
  const __bf16* __restrict__ bh0 = whi + (size_t)(ln +  0) * HID;
  const __bf16* __restrict__ bh1 = whi + (size_t)(ln + 16) * HID;
  const __bf16* __restrict__ bh2 = whi + (size_t)(ln + 32) * HID;
  const __bf16* __restrict__ bh3 = whi + (size_t)(ln + 48) * HID;
  const __bf16* __restrict__ bl0 = wlo + (size_t)(ln +  0) * HID;
  const __bf16* __restrict__ bl1 = wlo + (size_t)(ln + 16) * HID;
  const __bf16* __restrict__ bl2 = wlo + (size_t)(ln + 32) * HID;
  const __bf16* __restrict__ bl3 = wlo + (size_t)(ln + 48) * HID;

  v8f c0 = {}, c1 = {}, c2 = {}, c3 = {};

  for (int k = 0; k < HID; k += 32) {
    // A fragment (16x32 bf16): elems 0..15 -> K = half*16 + i
    v16bf ah, al;
    cvt8<0>(xrow + k + half * 16,     ah, al);
    cvt8<8>(xrow + k + half * 16 + 8, ah, al);

    // B fragments (32x16 bf16): elems 0..15 -> K = half*16 + i. Direct bf16 loads.
    const int kb = k + half * 16;
    v16bf b0h = *(const v16bf*)(bh0 + kb);
    v16bf b1h = *(const v16bf*)(bh1 + kb);
    v16bf b2h = *(const v16bf*)(bh2 + kb);
    v16bf b3h = *(const v16bf*)(bh3 + kb);
    v16bf b0l = *(const v16bf*)(bl0 + kb);
    v16bf b1l = *(const v16bf*)(bl1 + kb);
    v16bf b2l = *(const v16bf*)(bl2 + kb);
    v16bf b3l = *(const v16bf*)(bl3 + kb);

    // bf16x2 split accumulation: hi*hi + lo*hi + hi*lo (f32 accumulate)
    c0 = WMMA_BF16(c0, ah, b0h); c0 = WMMA_BF16(c0, al, b0h); c0 = WMMA_BF16(c0, ah, b0l);
    c1 = WMMA_BF16(c1, ah, b1h); c1 = WMMA_BF16(c1, al, b1h); c1 = WMMA_BF16(c1, ah, b1l);
    c2 = WMMA_BF16(c2, ah, b2h); c2 = WMMA_BF16(c2, al, b2h); c2 = WMMA_BF16(c2, ah, b2l);
    c3 = WMMA_BF16(c3, ah, b3h); c3 = WMMA_BF16(c3, al, b3h); c3 = WMMA_BF16(c3, ah, b3l);
  }

  // Spill logit tiles to LDS. C layout: VGPR r -> M = half*8 + r, N = ln.
  {
    const int r0 = wave * 16 + half * 8;
#pragma unroll
    for (int r = 0; r < 8; ++r) {
      float* row = &lds[(r0 + r) * LDS_STRIDE];
      row[ 0 + ln] = c0[r];
      row[16 + ln] = c1[r];
      row[32 + ln] = c2[r];
      row[48 + ln] = c3[r];
    }
  }
  __syncthreads();

  const size_t S = (size_t)TOKENS * NEXP;

  // Phase 2: one thread per row -> softplus, L1 normalize, top-2.
  if (tid < ROWS_PER_BLOCK) {
    const int row = blockIdx.x * ROWS_PER_BLOCK + tid;
    float* lrow = &lds[tid * LDS_STRIDE];
    float sum = 0.0f;
    float best0 = -1.0f, best1 = -1.0f;
    int   i0 = 0, i1 = 0;
    for (int e = 0; e < NEXP; ++e) {
      float lg = lrow[e];
      // numerically-stable softplus: log1p(exp(lg))
      float sp = fmaxf(lg, 0.0f) + log1pf(__expf(-fabsf(lg)));
      lrow[e] = sp;
      sum += sp;                         // softplus > 0 => sum == L1 norm
      if (sp > best0) { best1 = best0; i1 = i0; best0 = sp; i0 = e; }
      else if (sp > best1) { best1 = sp; i1 = e; }
    }
    float inv = 1.0f / fmaxf(sum, 1e-12f);
    float w0 = best0 * inv, w1 = best1 * inv;
    out[S + (size_t)row * 2 + 0] = w0;
    out[S + (size_t)row * 2 + 1] = w1;
    out[S + (size_t)TOKENS * 2 + (size_t)row * 2 + 0] = (float)i0;
    out[S + (size_t)TOKENS * 2 + (size_t)row * 2 + 1] = (float)i1;
    float g0 = 1.0f - w0, g1 = 1.0f - w1;
    lds_spk[tid] = g0 * g0 + g1 * g1;
    for (int e = 0; e < NEXP; ++e) lrow[e] *= inv;  // scores in LDS
  }
  __syncthreads();

  // Cooperative coalesced store of normalized scores.
  for (int idx = tid; idx < ROWS_PER_BLOCK * NEXP; idx += BLOCK_THREADS) {
    int r = idx >> 6, e = idx & 63;
    out[(size_t)(blockIdx.x * ROWS_PER_BLOCK + r) * NEXP + e] = lds[r * LDS_STRIDE + e];
  }

  // Block-local reductions for the loss terms -> one atomicAdd per expert per block.
  if (tid < NEXP) {
    float cs = 0.0f;
    for (int r = 0; r < ROWS_PER_BLOCK; ++r) cs += lds[r * LDS_STRIDE + tid];
    atomicAdd(&ws[tid], cs);
  }
  if (tid == 0) {
    float s = 0.0f;
    for (int r = 0; r < ROWS_PER_BLOCK; ++r) s += lds_spk[r];
    atomicAdd(&ws[NEXP], s);
  }
}

__global__ void finalize_kernel(const float* __restrict__ ws, float* __restrict__ out) {
  if (threadIdx.x != 0) return;
  const float uf = 1.0f / (float)NEXP;
  float bal = 0.0f;
  for (int e = 0; e < NEXP; ++e) {
    float d = ws[e] / (float)TOKENS - uf;
    bal += d * d;
  }
  bal *= uf;  // mean over experts
  const float maxval  = (1.0f - uf) * (1.0f - uf);
  const float zerov   = (float)(NEXP - 1) * uf * uf;
  const float max_bal = (maxval + zerov) * uf;
  bal /= max_bal;

  float spk = ws[NEXP] / ((float)TOKENS * 2.0f);
  const float min_spk = 0.25f;     // (1 - 1/2)^2
  const float max_spk = maxval;    // (1 - 1/64)^2
  spk = (spk - min_spk) / (max_spk - min_spk);

  out[(size_t)TOKENS * NEXP + (size_t)TOKENS * 4] = spk + bal;
}

extern "C" void kernel_launch(void* const* d_in, const int* in_sizes, int n_in,
                              void* d_out, int out_size, void* d_ws, size_t ws_size,
                              hipStream_t stream) {
  (void)in_sizes; (void)n_in; (void)out_size; (void)ws_size;
  const float* x = (const float*)d_in[0];
  const float* W = (const float*)d_in[1];
  float* out = (float*)d_out;
  float* ws  = (float*)d_ws;

  __bf16* whi = (__bf16*)((char*)d_ws + WS_RED_BYTES);
  __bf16* wlo = whi + W_ELEMS;

  zero_ws_kernel<<<1, 128, 0, stream>>>(ws);
  prep_w_kernel<<<W_ELEMS / 512, 256, 0, stream>>>(W, whi, wlo);
  router_kernel<<<TOKENS / ROWS_PER_BLOCK, BLOCK_THREADS, 0, stream>>>(x, whi, wlo, out, ws);
  finalize_kernel<<<1, 32, 0, stream>>>(ws, out);
}
